// DeformableSatAttention_85968065397291
// MI455X (gfx1250) — compile-verified, matrix operand order checked by reading, not run on hardware
//
#include <hip/hip_runtime.h>
#include <hip/hip_bf16.h>

// ---------------------------------------------------------------------------
// Multi-scale deformable attention for MI455X (gfx1250, wave32).
//   Stage 1: three GEMM+bias projections via v_wmma_f32_16x16x32_bf16
//            (4 M-tiles per wave -> each B fragment feeds 4 WMMAs)
//   Stage 2: wave-per-(b,q,head) softmax + bilinear gather (lane = channel)
// ---------------------------------------------------------------------------

typedef __bf16 v16bf __attribute__((ext_vector_type(16)));
typedef float  v8f   __attribute__((ext_vector_type(8)));

#define BATCH   2
#define NQ      5000
#define NV      13294
#define DMODEL  256
#define HEADS   8
#define LVLS    4
#define PTS     8
#define NREF    4
#define DH      32
#define KDIM    256   // inner dim of all three GEMMs
#define MT      4     // M-tiles (of 16 rows) per wave -> 64-row strip

// ---------------------------------------------------------------------------
// GEMM: C[M,N] = A[M,K=256] (f32, cvt->bf16) @ W[K,N] (f32, cvt->bf16) + bias
// Each wave owns a 64x16 strip: per 32-K chunk it builds ONE B fragment
// (16 strided gathers down a W column) and reuses it across 4 WMMAs whose A
// fragments come from coalesced row loads. Fragment layouts per CDNA5 ISA
// 7.12.2 (16-bit A 16x32 / B 32x16 / f32 C/D 16x16).
// ---------------------------------------------------------------------------
template <int N>
__global__ __launch_bounds__(256)
void gemm_bias_bf16_wmma(const float* __restrict__ A,
                         const float* __restrict__ W,
                         const float* __restrict__ bias,
                         float* __restrict__ C,
                         int M, int numTiles)
{
    const int wave = threadIdx.x >> 5;
    const int lane = threadIdx.x & 31;
    const int tile = blockIdx.x * 8 + wave;
    if (tile >= numTiles) return;                 // wave-uniform exit

    const int ntilesN = N / 16;
    const int m0 = (tile / ntilesN) * (16 * MT);  // 64-row strip base
    const int n0 = (tile % ntilesN) * 16;

    const int col   = n0 + (lane & 15);
    const int khalf = (lane >> 4) * 8;            // lanes 16-31 own K+8 half

    // clamped row index per sub-tile (loads stay in bounds; stores guarded)
    int row[MT];
#pragma unroll
    for (int i = 0; i < MT; ++i) {
        int rr = m0 + i * 16 + (lane & 15);
        row[i] = rr < M ? rr : M - 1;
    }

    v8f acc[MT];
#pragma unroll
    for (int i = 0; i < MT; ++i) acc[i] = v8f{};

#pragma unroll
    for (int k0 = 0; k0 < KDIM; k0 += 32) {
        // ---- B fragment: one column walk, reused by all MT WMMAs ----
        v16bf bfrag;
#pragma unroll
        for (int j = 0; j < 16; ++j) {
            const int k = k0 + khalf + (j & 7) + ((j & 8) ? 16 : 0);
            bfrag[j] = (__bf16)W[(size_t)k * N + col];
        }
        // prefetch next A chunk rows into cache (global_prefetch_b8)
        if (k0 + 32 < KDIM) {
#pragma unroll
            for (int i = 0; i < MT; ++i)
                __builtin_prefetch(&A[(size_t)row[i] * KDIM + k0 + 32], 0, 1);
        }
        // ---- MT A fragments (coalesced b128 row loads) + WMMAs ----
#pragma unroll
        for (int i = 0; i < MT; ++i) {
            v16bf afrag;
#pragma unroll
            for (int j = 0; j < 16; ++j) {
                const int k = k0 + khalf + (j & 7) + ((j & 8) ? 16 : 0);
                afrag[j] = (__bf16)A[(size_t)row[i] * KDIM + k];
            }
            acc[i] = __builtin_amdgcn_wmma_f32_16x16x32_bf16(
                         false, afrag, false, bfrag, (short)0, acc[i], false, false);
        }
    }

    const float bv = bias[col];
#pragma unroll
    for (int i = 0; i < MT; ++i) {
#pragma unroll
        for (int r = 0; r < 8; ++r) {
            const int mr = m0 + i * 16 + r + ((lane >> 4) << 3); // M = r (+8 hi lanes)
            if (mr < M)
                C[(size_t)mr * N + n0 + (lane & 15)] = acc[i][r] + bv;
        }
    }
}

// ---------------------------------------------------------------------------
// Sampling: one wave32 per (b, q, head). lane == channel (DH==32) so every
// corner gather is a single coalesced 128B L2 transaction.
// lane s also owns sample s = lvl*8+pt for softmax / location math; per-sample
// scalars are wave-broadcast with __shfl (ds_bpermute_b32).
// ---------------------------------------------------------------------------
__global__ __launch_bounds__(256)
void ms_deform_sample(const float* __restrict__ v,         // [B*NV, 256]
                      const float* __restrict__ off_log,   // [B*NQ, 512]
                      const float* __restrict__ attn_log,  // [B*NQ, 256]
                      const float* __restrict__ refpts,    // [B*NQ, NREF, 2]
                      float* __restrict__ out)             // [B*NQ, 256]
{
    const int Hs[LVLS] = {100, 50, 25, 13};
    const int Ws[LVLS] = {100, 50, 25, 13};
    const int St[LVLS] = {0, 10000, 12500, 13125};

    const int wave = threadIdx.x >> 5;
    const int lane = threadIdx.x & 31;
    const int widx = blockIdx.x * 8 + wave;
    if (widx >= BATCH * NQ * HEADS) return;       // wave-uniform exit

    const int b   = widx / (NQ * HEADS);
    const int rem = widx % (NQ * HEADS);
    const int q   = rem / HEADS;
    const int h   = rem % HEADS;
    const size_t qrow = (size_t)(b * NQ + q);

    // ---- per-lane sample s = lane : lvl = s>>3, pt = s&7 ----
    const int l = lane >> 3;
    const int p = lane & 7;

    // softmax over the 32 (lvl,pt) logits of this head (5-step shfl_xor tree)
    float logit = attn_log[qrow * 256 + h * 32 + lane];
    float mx = logit;
#pragma unroll
    for (int o = 16; o; o >>= 1) mx = fmaxf(mx, __shfl_xor(mx, o, 32));
    float e = __expf(logit - mx);
    float sm = e;
#pragma unroll
    for (int o = 16; o; o >>= 1) sm += __shfl_xor(sm, o, 32);
    const float aw = e / sm;

    // sampling location for this lane's sample
    const int   Hl = Hs[l], Wl = Ws[l];
    const size_t ob = qrow * 512 + (size_t)h * 64 + l * 16 + p * 2;
    const float ox = off_log[ob], oy = off_log[ob + 1];
    const int   r  = p & (NREF - 1);              // pt -> reference point
    const size_t rb = (qrow * NREF + r) * 2;
    const float lx = refpts[rb]     + ox / (float)Wl;
    const float ly = refpts[rb + 1] + oy / (float)Hl;
    const float x  = lx * (float)Wl - 0.5f;
    const float y  = ly * (float)Hl - 0.5f;
    const float fx0 = floorf(x), fy0 = floorf(y);
    const float dx = x - fx0,  dy = y - fy0;
    const int   x0 = (int)fx0, y0 = (int)fy0;

    const float* vb = v + (size_t)(b * NV) * 256 + h * 32 + lane;
    float acc = 0.f;

#pragma unroll 4
    for (int s = 0; s < 32; ++s) {
        const int   sx0 = __shfl(x0, s, 32);
        const int   sy0 = __shfl(y0, s, 32);
        const float sdx = __shfl(dx, s, 32);
        const float sdy = __shfl(dy, s, 32);
        const float sw  = __shfl(aw, s, 32);
        const int   sl  = s >> 3;
        const int   HH = Hs[sl], WW = Ws[sl], ST = St[sl];

        const int x1 = sx0 + 1, y1 = sy0 + 1;
        const bool vx0 = (sx0 >= 0) & (sx0 < WW);
        const bool vx1 = (x1  >= 0) & (x1  < WW);
        const bool vy0 = (sy0 >= 0) & (sy0 < HH);
        const bool vy1 = (y1  >= 0) & (y1  < HH);
        const int cx0 = min(max(sx0, 0), WW - 1);
        const int cx1 = min(max(x1,  0), WW - 1);
        const int cy0 = min(max(sy0, 0), HH - 1);
        const int cy1 = min(max(y1,  0), HH - 1);

        const float g00 = (vx0 && vy0) ? vb[(size_t)(ST + cy0 * WW + cx0) * 256] : 0.f;
        const float g01 = (vx1 && vy0) ? vb[(size_t)(ST + cy0 * WW + cx1) * 256] : 0.f;
        const float g10 = (vx0 && vy1) ? vb[(size_t)(ST + cy1 * WW + cx0) * 256] : 0.f;
        const float g11 = (vx1 && vy1) ? vb[(size_t)(ST + cy1 * WW + cx1) * 256] : 0.f;

        const float w00 = (1.f - sdx) * (1.f - sdy);
        const float w01 = sdx * (1.f - sdy);
        const float w10 = (1.f - sdx) * sdy;
        const float w11 = sdx * sdy;
        acc += sw * (g00 * w00 + g01 * w01 + g10 * w10 + g11 * w11);
    }

    out[qrow * 256 + h * 32 + lane] = acc;
}

// ---------------------------------------------------------------------------
// Host launch
// ---------------------------------------------------------------------------
extern "C" void kernel_launch(void* const* d_in, const int* in_sizes, int n_in,
                              void* d_out, int out_size, void* d_ws, size_t ws_size,
                              hipStream_t stream)
{
    (void)in_sizes; (void)n_in; (void)out_size; (void)ws_size;

    const float* query    = (const float*)d_in[0];   // [B,NQ,256]
    const float* value    = (const float*)d_in[2];   // [B,NV,256]
    const float* refpts   = (const float*)d_in[3];   // [B,NQ,NREF,2]
    const float* W_value  = (const float*)d_in[6];   // [256,256]
    const float* b_value  = (const float*)d_in[7];   // [256]
    const float* W_offset = (const float*)d_in[8];   // [256,512]
    const float* b_offset = (const float*)d_in[9];   // [512]
    const float* W_attn   = (const float*)d_in[10];  // [256,256]
    const float* b_attn   = (const float*)d_in[11];  // [256]
    float*       out      = (float*)d_out;

    const int MV = BATCH * NV;   // 26588 rows of value projection
    const int MQ = BATCH * NQ;   // 10000 rows of query projections

    // workspace layout (f32): v | off_logits | attn_logits  (~58 MB)
    float* vproj  = (float*)d_ws;
    float* offlog = vproj  + (size_t)MV * 256;
    float* attnlg = offlog + (size_t)MQ * 512;

    const int STRIP = 16 * MT;   // 64 rows per wave

    // value projection
    {
        const int tiles = ((MV + STRIP - 1) / STRIP) * (DMODEL / 16);
        gemm_bias_bf16_wmma<DMODEL><<<(tiles + 7) / 8, 256, 0, stream>>>(
            value, W_value, b_value, vproj, MV, tiles);
    }
    // offset projection: N=512
    {
        const int tiles = ((MQ + STRIP - 1) / STRIP) * (512 / 16);
        gemm_bias_bf16_wmma<512><<<(tiles + 7) / 8, 256, 0, stream>>>(
            query, W_offset, b_offset, offlog, MQ, tiles);
    }
    // attention-logit projection: N=256
    {
        const int tiles = ((MQ + STRIP - 1) / STRIP) * (DMODEL / 16);
        gemm_bias_bf16_wmma<DMODEL><<<(tiles + 7) / 8, 256, 0, stream>>>(
            query, W_attn, b_attn, attnlg, MQ, tiles);
    }
    // sampling: one wave per (b,q,head) -> 80000 waves, 8 waves/block
    {
        const int waves = BATCH * NQ * HEADS;
        ms_deform_sample<<<(waves + 7) / 8, 256, 0, stream>>>(
            vproj, offlog, attnlg, refpts, out);
    }
}
